// InternS1ProMoeSparseMoeBlock_83597243449695
// MI455X (gfx1250) — matrix operand order repaired and checked by reading, not compiled, not run on hardware
//
#include <hip/hip_runtime.h>
#include <hip/hip_bf16.h>
#include <math.h>

// ---------------- problem constants ----------------
constexpr int kT = 2048;   // tokens
constexpr int kH = 1024;   // hidden
constexpr int kE = 64;     // experts
constexpr int kI = 512;    // intermediate
constexpr int kK = 8;      // top-k
constexpr int kG = 4;      // groups
constexpr int kRowsMax = kT * kK + kE * 16;   // 17408 (16-aligned per-expert segments)
constexpr int kNTiles  = kRowsMax / 16;       // 1088 row tiles (upper bound)
constexpr int kXStride = kH + 8;              // padded LDS row stride (bf16 elems)

typedef __attribute__((ext_vector_type(16))) __bf16 v16bf;
typedef __attribute__((ext_vector_type(8)))  __bf16 v8bf;
typedef __attribute__((ext_vector_type(4)))  __bf16 v4bf;
typedef __attribute__((ext_vector_type(8)))  float  v8f;
typedef __attribute__((ext_vector_type(4)))  unsigned int v4u;
typedef __attribute__((ext_vector_type(2)))  unsigned int v2u;

static __device__ __forceinline__ v16bf concat8(v8bf lo, v8bf hi) {
    return __builtin_shufflevector(lo, hi, 0, 1, 2, 3, 4, 5, 6, 7,
                                   8, 9, 10, 11, 12, 13, 14, 15);
}

// ---------------- init: zero output, pair arrays, counters ----------------
__global__ void moe_init_kernel(float* __restrict__ out, int* __restrict__ pair_tok,
                                float* __restrict__ pair_w, int* __restrict__ counts,
                                int* __restrict__ fill) {
    int i = blockIdx.x * blockDim.x + threadIdx.x;
    int stride = gridDim.x * blockDim.x;
    for (int j = i; j < kT * kH; j += stride) out[j] = 0.f;
    for (int j = i; j < kRowsMax; j += stride) { pair_tok[j] = -1; pair_w[j] = 0.f; }
    if (i < kE) { counts[i] = 0; fill[i] = 0; }
}

// ---------------- router: logits -> exp(max-sub) -> group top-2 -> renorm ----------------
__global__ __launch_bounds__(64) void moe_router_kernel(
    const float* __restrict__ x, const float* __restrict__ gw,
    float* __restrict__ topk_w, int* __restrict__ topk_id, int* __restrict__ counts) {
    __shared__ float xs[kH];
    __shared__ float p[kE];
    __shared__ float tw[kK];
    __shared__ int   ti[kK];
    __shared__ float s_sum;

    int t = blockIdx.x;
    for (int i = threadIdx.x; i < kH; i += 64) xs[i] = x[(size_t)t * kH + i];
    __syncthreads();

    int e = threadIdx.x;  // 0..63, one expert per thread
    const float* w = gw + (size_t)e * kH;
    float acc = 0.f;
    for (int i = 0; i < kH; i += 4)
        acc += xs[i] * w[i] + xs[i + 1] * w[i + 1] + xs[i + 2] * w[i + 2] + xs[i + 3] * w[i + 3];
    p[e] = acc;
    __syncthreads();

    float m = p[0];
    for (int j = 1; j < kE; ++j) m = fmaxf(m, p[j]);
    __syncthreads();
    // softmax denominator cancels under top-k selection + renormalization; keep exp only
    p[e] = __expf(acc - m);
    __syncthreads();

    if (threadIdx.x < kG) {
        int g = threadIdx.x, base = g * (kE / kG);
        float b1 = -1.f, b2 = -1.f; int i1 = 0, i2 = 0;
        for (int j = 0; j < kE / kG; ++j) {
            float v = p[base + j];
            if (v > b1)      { b2 = b1; i2 = i1; b1 = v; i1 = j; }
            else if (v > b2) { b2 = v; i2 = j; }
        }
        tw[g * 2]     = b1; ti[g * 2]     = base + i1;
        tw[g * 2 + 1] = b2; ti[g * 2 + 1] = base + i2;
    }
    __syncthreads();
    if (threadIdx.x == 0) {
        float s = 0.f;
        for (int k = 0; k < kK; ++k) s += tw[k];
        s_sum = s;
    }
    __syncthreads();
    if (threadIdx.x < kK) {
        int k = threadIdx.x;
        topk_w[(size_t)t * kK + k]  = tw[k] / s_sum;
        topk_id[(size_t)t * kK + k] = ti[k];
        atomicAdd(&counts[ti[k]], 1);
    }
}

// ---------------- offsets: 16-aligned segment prefix sum + tile table ----------------
__global__ void moe_offsets_kernel(const int* __restrict__ counts, int* __restrict__ offsets,
                                   int* __restrict__ tile_e, int* __restrict__ tile_r0) {
    if (threadIdx.x == 0 && blockIdx.x == 0) {
        int off = 0, tile = 0;
        for (int e = 0; e < kE; ++e) {
            offsets[e] = off;
            int nt = (counts[e] + 15) >> 4;
            for (int tt = 0; tt < nt && tile < kNTiles; ++tt) {
                tile_e[tile]  = e;
                tile_r0[tile] = off + tt * 16;
                ++tile;
            }
            off += nt * 16;
        }
        offsets[kE] = off;
        for (; tile < kNTiles; ++tile) { tile_e[tile] = -1; tile_r0[tile] = 0; }
    }
}

// ---------------- scatter pairs into compacted per-expert segments ----------------
__global__ __launch_bounds__(256) void moe_scatter_kernel(
    const int* __restrict__ topk_id, const float* __restrict__ topk_w,
    const int* __restrict__ offsets, int* __restrict__ fill,
    int* __restrict__ pair_tok, float* __restrict__ pair_w) {
    int i = blockIdx.x * blockDim.x + threadIdx.x;  // over T*K pairs
    if (i < kT * kK) {
        int e   = topk_id[i];
        int pos = atomicAdd(&fill[e], 1);
        int row = offsets[e] + pos;
        pair_tok[row] = i / kK;
        pair_w[row]   = topk_w[i];
    }
}

// ---------------- GEMM1 + SwiGLU: h = silu(x@Wg^T) * (x@Wu^T) ----------------
// block: 256 threads (8 waves). waves 0-3: gate 16x16 tiles, waves 4-7: up tiles.
// grid: (kNTiles, kI/64). Whole 16xH x-tile staged once in LDS (bf16, ~33KB);
// K loop is barrier-free: 2x ds_load_b128 (A) + 4x global_load_b128 (B) + wmma.
__global__ __launch_bounds__(256) void moe_gemm1_kernel(
    const float* __restrict__ x, const float* __restrict__ w13,
    const int* __restrict__ tile_e, const int* __restrict__ tile_r0,
    const int* __restrict__ pair_tok, __bf16* __restrict__ hbuf) {
    int tile = blockIdx.x;
    int e = tile_e[tile];
    if (e < 0) return;  // uniform per-block exit
    int row0 = tile_r0[tile];
    int c0   = blockIdx.y * 64;            // i-channel base
    int wave = threadIdx.x >> 5;
    int lane = threadIdx.x & 31;
    bool isUp = wave >= 4;
    int wquad = wave & 3;
    int n  = lane & 15;
    int kb = (lane >> 4) * 8;              // 16-bit A layout: lanes 16-31 hold K+8

    __shared__ int s_tok[16];
    __shared__ __align__(16) __bf16 xa[16][kXStride];   // full 16 x H tile
    __shared__ float utile[4][16][16];
    __shared__ __align__(16) __bf16 ht[16][72];         // staged h tile (16x64)

    if (threadIdx.x < 16) s_tok[threadIdx.x] = pair_tok[row0 + threadIdx.x];
    __syncthreads();

    // stage the whole x tile: iteration q loads row q fully coalesced (256 x float4)
    #pragma unroll 4
    for (int i = threadIdx.x; i < 16 * (kH / 4); i += 256) {
        int r  = i >> 8;               // row (kH/4 == 256 float4 chunks per row)
        int c4 = (i & 255) * 4;        // column base
        int tok = s_tok[r];
        float4 f = (tok >= 0) ? *(const float4*)(x + (size_t)tok * kH + c4)
                              : make_float4(0.f, 0.f, 0.f, 0.f);
        v4bf h;
        h[0] = (__bf16)f.x; h[1] = (__bf16)f.y; h[2] = (__bf16)f.z; h[3] = (__bf16)f.w;
        *(v2u*)&xa[r][c4] = __builtin_bit_cast(v2u, h);
    }
    __syncthreads();

    // w13 row index (gate rows [0,I), up rows [I,2I))
    int col = (isUp ? kI : 0) + c0 + 16 * wquad + n;
    const float* bsrc = w13 + ((size_t)e * (2 * kI) + col) * kH + 8 * (lane >> 4);

    v8f acc = {};
    #pragma unroll 2
    for (int k0 = 0; k0 < kH; k0 += 32) {
        // A fragment: row = lane&15, two contiguous 16B chunks at kb and kb+16
        v8bf alo = __builtin_bit_cast(v8bf, *(const v4u*)&xa[n][k0 + kb]);
        v8bf ahi = __builtin_bit_cast(v8bf, *(const v4u*)&xa[n][k0 + 16 + kb]);
        v16bf a = concat8(alo, ahi);
        // B fragment: two runs of 8 f32 along H at kb and kb+16, convert to bf16
        v16bf b;
        const float* bp = bsrc + k0;
        __builtin_prefetch(bp + 64, 0, 1);   // prefetch ahead -> global_prefetch
        #pragma unroll
        for (int j = 0; j < 16; ++j) b[j] = (__bf16)bp[(j & 7) + ((j >> 3) << 4)];

        acc = __builtin_amdgcn_wmma_f32_16x16x32_bf16(false, a, false, b, (short)0, acc,
                                                      false, false);
    }

    // exchange up tiles -> gate waves fuse SiLU and stage bf16 h tile in LDS
    int mb = (lane >> 4) * 8;
    if (isUp) {
        #pragma unroll
        for (int r = 0; r < 8; ++r) utile[wquad][mb + r][n] = acc[r];
    }
    __syncthreads();
    if (!isUp) {
        #pragma unroll
        for (int r = 0; r < 8; ++r) {
            float g = acc[r];
            float u = utile[wquad][mb + r][n];
            float hv = (g / (1.f + __expf(-g))) * u;
            ht[mb + r][16 * wquad + n] = (__bf16)hv;
        }
    }
    __syncthreads();
    // coalesced writeback: one 8B store per thread (16 rows x 64 cols bf16)
    {
        int r  = threadIdx.x >> 4;
        int cc = (threadIdx.x & 15) * 4;
        v4bf hv = __builtin_bit_cast(v4bf, *(const v2u*)&ht[r][cc]);
        *(v2u*)&hbuf[(size_t)(row0 + r) * kI + c0 + cc] = __builtin_bit_cast(v2u, hv);
    }
}

// ---------------- GEMM2 + weighted scatter-add: out[tok] += w * (h @ w2[e]^T) ----------------
// block: 256 threads (8 waves), each wave one 16x16 output tile; grid (kNTiles, kH/128)
__global__ __launch_bounds__(256) void moe_gemm2_kernel(
    const __bf16* __restrict__ hbuf, const float* __restrict__ w2,
    const int* __restrict__ tile_e, const int* __restrict__ tile_r0,
    const int* __restrict__ pair_tok, const float* __restrict__ pair_w,
    float* __restrict__ out) {
    int tile = blockIdx.x;
    int e = tile_e[tile];
    if (e < 0) return;
    int row0 = tile_r0[tile];
    int wave = threadIdx.x >> 5;
    int lane = threadIdx.x & 31;
    int n  = lane & 15;
    int kb = (lane >> 4) * 8;
    int hcol = blockIdx.y * 128 + wave * 16 + n;

    __shared__ int   s_tok[16];
    __shared__ float s_w[16];
    if (threadIdx.x < 16) {
        s_tok[threadIdx.x] = pair_tok[row0 + threadIdx.x];
        s_w[threadIdx.x]   = pair_w[row0 + threadIdx.x];
    }
    __syncthreads();

    const __bf16* arow = hbuf + (size_t)(row0 + n) * kI + kb;         // A: row = lane&15
    const float*  bsrc = w2 + ((size_t)e * kH + hcol) * kI + kb;      // 8 f32 at kb, 8 at kb+16

    v8f acc = {};
    #pragma unroll 4
    for (int k0 = 0; k0 < kI; k0 += 32) {
        const __bf16* ap = arow + k0;
        v8bf alo = __builtin_bit_cast(v8bf, *(const v4u*)(ap));
        v8bf ahi = __builtin_bit_cast(v8bf, *(const v4u*)(ap + 16));
        v16bf a = concat8(alo, ahi);
        v16bf b;
        const float* bp = bsrc + k0;
        __builtin_prefetch(bp + 64, 0, 1);
        #pragma unroll
        for (int j = 0; j < 16; ++j) b[j] = (__bf16)bp[(j & 7) + ((j >> 3) << 4)];
        acc = __builtin_amdgcn_wmma_f32_16x16x32_bf16(false, a, false, b, (short)0, acc,
                                                      false, false);
    }

    int mb = (lane >> 4) * 8;
    #pragma unroll
    for (int r = 0; r < 8; ++r) {
        int m = mb + r;
        int tok = s_tok[m];
        if (tok >= 0)
            atomicAdd(&out[(size_t)tok * kH + hcol], acc[r] * s_w[m]);
    }
}

// ---------------- launch ----------------
extern "C" void kernel_launch(void* const* d_in, const int* in_sizes, int n_in,
                              void* d_out, int out_size, void* d_ws, size_t ws_size,
                              hipStream_t stream) {
    (void)in_sizes; (void)n_in; (void)out_size; (void)ws_size;
    const float* x   = (const float*)d_in[0];  // [T, H]
    const float* gw  = (const float*)d_in[1];  // [E, H]
    const float* w13 = (const float*)d_in[2];  // [E, 2I, H]
    const float* w2  = (const float*)d_in[3];  // [E, H, I]
    float* out = (float*)d_out;                // [T, H]

    char* ws = (char*)d_ws;
    size_t off = 0;
    auto carve = [&](size_t bytes) -> void* {
        void* p = ws + off;
        off = (off + bytes + 255) & ~(size_t)255;
        return p;
    };
    float* topk_w   = (float*)carve((size_t)kT * kK * sizeof(float));
    int*   topk_id  = (int*)  carve((size_t)kT * kK * sizeof(int));
    int*   counts   = (int*)  carve(kE * sizeof(int));
    int*   fill     = (int*)  carve(kE * sizeof(int));
    int*   offsets  = (int*)  carve((kE + 1) * sizeof(int));
    int*   tile_e   = (int*)  carve(kNTiles * sizeof(int));
    int*   tile_r0  = (int*)  carve(kNTiles * sizeof(int));
    int*   pair_tok = (int*)  carve(kRowsMax * sizeof(int));
    float* pair_w   = (float*)carve(kRowsMax * sizeof(float));
    __bf16* hbuf    = (__bf16*)carve((size_t)kRowsMax * kI * sizeof(__bf16));  // ~17.8 MB

    moe_init_kernel<<<2048, 256, 0, stream>>>(out, pair_tok, pair_w, counts, fill);
    moe_router_kernel<<<kT, 64, 0, stream>>>(x, gw, topk_w, topk_id, counts);
    moe_offsets_kernel<<<1, 32, 0, stream>>>(counts, offsets, tile_e, tile_r0);
    moe_scatter_kernel<<<(kT * kK) / 256, 256, 0, stream>>>(topk_id, topk_w, offsets, fill,
                                                            pair_tok, pair_w);
    dim3 g1(kNTiles, kI / 64);
    moe_gemm1_kernel<<<g1, 256, 0, stream>>>(x, w13, tile_e, tile_r0, pair_tok, hbuf);
    dim3 g2(kNTiles, kH / 128);
    moe_gemm2_kernel<<<g2, 256, 0, stream>>>(hbuf, w2, tile_e, tile_r0, pair_tok, pair_w, out);
}
